// GraphMoeV13ConfidenceGate_72267119722662
// MI455X (gfx1250) — matrix operand order repaired and checked by reading, not compiled, not run on hardware
//
#include <hip/hip_runtime.h>
#include <hip/hip_bf16.h>

// ---------------------------------------------------------------------------
// GraphMoE (top-2 of 8 experts, confidence-gated residual) for MI455X/gfx1250.
//
// - All GEMMs via v_wmma_f32_16x16x32_bf16 (f32 accumulate).
// - Top-2 sparsity: per-expert compaction lists -> grouped GEMMs (4x FLOP cut
//   vs dense expert eval); conf folded into gates; expert outputs accumulate
//   straight into h with global f32 atomics.
// - Aggregation via launch-built CSR: pure gather-reduce, zero per-layer
//   atomics, normalize fused in.
// - Weights pre-swizzled into the native WMMA B-fragment layout (one
//   contiguous 32B load per lane per fragment).
// - Weak experts process M=32 per wave: each B fragment feeds 2 WMMAs.
// ---------------------------------------------------------------------------

typedef __attribute__((ext_vector_type(16))) __bf16 v16bf;
typedef __attribute__((ext_vector_type(8)))  float  v8f;

#define HDIM   256
#define INDIM  128
#define OUTDIM 128
#define NLAYER 3
#define NEXP   8   // 4 weak + 4 strong

__device__ __forceinline__ unsigned short f2bf(float f) {
    union { float f; unsigned u; } x; x.f = f;
    unsigned u = x.u;
    unsigned r = u + 0x7FFFu + ((u >> 16) & 1u);
    return (unsigned short)(r >> 16);
}

// A-fragment: two contiguous 16B chunks (K = 8*half and 16+8*half in-tile).
__device__ __forceinline__ v16bf ld2(const unsigned short* p0, const unsigned short* p1) {
    union { v16bf v; uint4 q[2]; } u;
    u.q[0] = *reinterpret_cast<const uint4*>(p0);
    u.q[1] = *reinterpret_cast<const uint4*>(p1);
    return u.v;
}

// Pre-swizzled B fragment: one contiguous 32B load.
__device__ __forceinline__ v16bf ldfrag(const unsigned short* p) {
    union { v16bf v; uint4 q[2]; } u;
    const uint4* q = reinterpret_cast<const uint4*>(p);
    u.q[0] = q[0]; u.q[1] = q[1];
    return u.v;
}

__device__ __forceinline__ v8f wmma_bf16(v16bf a, v16bf b, v8f c) {
    return __builtin_amdgcn_wmma_f32_16x16x32_bf16(false, a, false, b, (short)0, c, false, false);
}

// ---------------------------------------------------------------------------
// utility kernels
// ---------------------------------------------------------------------------
__global__ void zero_kernel(float* p, long long n) {
    long long i = (long long)blockIdx.x * 256 + threadIdx.x;
    if (i < n) p[i] = 0.0f;
}

__global__ void cvt_bf16_kernel(const float* in, unsigned short* out, long long n) {
    long long i = (long long)blockIdx.x * 256 + threadIdx.x;
    if (i < n) out[i] = f2bf(in[i]);
}

// [K,Ncols] f32 (nmats contiguous) -> bf16 B-fragment order:
// frag(mat,kt,nt)[lane][i]: K = kt*32 + 8*(lane>=16) + (i&7) + 16*(i>=8), N = nt*16 + (lane&15)
__global__ __launch_bounds__(32) void swizzleB_kernel(const float* W, unsigned short* out,
                                                      int K, int Ncols) {
    int KT = K >> 5, NT = Ncols >> 4;
    int per = KT * NT;
    int b = blockIdx.x;
    int mat = b / per, r = b % per;
    int kt = r / NT, nt = r % NT;
    int lane = threadIdx.x;
    const float* Wm = W + (size_t)mat * K * Ncols;
    unsigned short* op = out + ((size_t)b * 32 + lane) * 16;
    int n = nt * 16 + (lane & 15);
    int kbase = kt * 32 + ((lane >> 4) << 3);
#pragma unroll
    for (int i = 0; i < 16; ++i)
        op[i] = f2bf(Wm[(size_t)(kbase + (i & 7) + ((i >> 3) << 4)) * Ncols + n]);
}

// ---------------------------------------------------------------------------
// CSR construction (once per launch; edge_index is launch-invariant)
// ---------------------------------------------------------------------------
__global__ void hist_kernel(const long long* ei, int* degi, int E) {
    int e = blockIdx.x * 256 + threadIdx.x;
    if (e < E) atomicAdd(degi + (int)ei[(size_t)E + e], 1);
}

// single-block exclusive scan: rowptr[0..N] from degi[0..N-1]
__global__ __launch_bounds__(256) void scan_kernel(const int* degi, int* rowptr, int N) {
    __shared__ int part[256];
    int t = threadIdx.x;
    int chunk = (N + 255) / 256;
    int b0 = t * chunk;
    int b1 = b0 + chunk; if (b1 > N) b1 = N;
    int sum = 0;
    for (int i = b0; i < b1 && i < N; ++i) sum += degi[i];
    part[t] = sum;
    __syncthreads();
    if (t == 0) {
        int acc = 0;
        for (int i = 0; i < 256; ++i) { int v = part[i]; part[i] = acc; acc += v; }
    }
    __syncthreads();
    int acc = part[t];
    for (int i = b0; i < b1 && i < N; ++i) { rowptr[i] = acc; acc += degi[i]; }
    if (b1 >= N && b0 <= N) rowptr[N] = acc;   // duplicate same-value writes: benign
}

__global__ void copy_int_kernel(const int* in, int* out, int n) {
    int i = blockIdx.x * 256 + threadIdx.x;
    if (i < n) out[i] = in[i];
}

__global__ void build_csr_kernel(const long long* ei, int* fill, int* csr, int E) {
    int e = blockIdx.x * 256 + threadIdx.x;
    if (e >= E) return;
    int dst = (int)ei[(size_t)E + e];
    int pos = atomicAdd(fill + dst, 1);
    csr[pos] = (int)ei[e];
}

// ---------------------------------------------------------------------------
// aggregation: one wave per node. agg[n] = mean over CSR neighbors of h[src].
// Emits f32 (router) + bf16 (WMMA A) copies. No atomics, normalize fused.
// ---------------------------------------------------------------------------
__global__ __launch_bounds__(256) void aggregate_kernel(const float* h, const int* csr,
                                                        const int* rowptr, float* agg,
                                                        unsigned short* aggb, int N) {
    int wid = threadIdx.x >> 5, lane = threadIdx.x & 31;
    int n = blockIdx.x * 8 + wid;
    if (n >= N) return;
    int s = rowptr[n], e = rowptr[n + 1];
    float a0x = 0, a0y = 0, a0z = 0, a0w = 0;
    float a1x = 0, a1y = 0, a1z = 0, a1w = 0;
    for (int j = s; j < e; ++j) {
        int src = csr[j];
        const float* hr = h + (size_t)src * HDIM + lane * 4;
        float4 v0 = *reinterpret_cast<const float4*>(hr);
        float4 v1 = *reinterpret_cast<const float4*>(hr + 128);
        a0x += v0.x; a0y += v0.y; a0z += v0.z; a0w += v0.w;
        a1x += v1.x; a1y += v1.y; a1z += v1.z; a1w += v1.w;
    }
    int deg = e - s;
    float sc = 1.0f / (float)(deg < 1 ? 1 : deg);
    a0x *= sc; a0y *= sc; a0z *= sc; a0w *= sc;
    a1x *= sc; a1y *= sc; a1z *= sc; a1w *= sc;
    float* ap = agg + (size_t)n * HDIM + lane * 4;
    *reinterpret_cast<float4*>(ap)       = make_float4(a0x, a0y, a0z, a0w);
    *reinterpret_cast<float4*>(ap + 128) = make_float4(a1x, a1y, a1z, a1w);
    uint2 p0, p1;
    p0.x = (unsigned)f2bf(a0x) | ((unsigned)f2bf(a0y) << 16);
    p0.y = (unsigned)f2bf(a0z) | ((unsigned)f2bf(a0w) << 16);
    p1.x = (unsigned)f2bf(a1x) | ((unsigned)f2bf(a1y) << 16);
    p1.y = (unsigned)f2bf(a1z) | ((unsigned)f2bf(a1w) << 16);
    unsigned short* bp = aggb + (size_t)n * HDIM + lane * 4;
    *reinterpret_cast<uint2*>(bp)       = p0;
    *reinterpret_cast<uint2*>(bp + 128) = p1;
}

// ---------------------------------------------------------------------------
// router: logits = [h,agg] @ router_W + b, top-2 softmax, conf = sigmoid(...)
// one wave per node; appends (node, gate*conf) to per-expert lists
// ---------------------------------------------------------------------------
__global__ __launch_bounds__(256) void router_kernel(const float* h, const float* agg,
                                                     const float* rW, const float* rb,
                                                     const float* cW, const float* cb,
                                                     int* counts, int* idxl, float* wl,
                                                     int l, int N) {
    int wid = threadIdx.x >> 5, lane = threadIdx.x & 31;
    int n = blockIdx.x * 8 + wid;
    if (n >= N) return;
    const float* hn = h + (size_t)n * HDIM;
    const float* an = agg + (size_t)n * HDIM;
    const float* rwl = rW + (size_t)l * (2 * HDIM) * NEXP;
    float acc[NEXP];
#pragma unroll
    for (int e = 0; e < NEXP; ++e) acc[e] = 0.0f;
    for (int t = 0; t < 16; ++t) {
        int d = lane + t * 32;
        float v = (d < HDIM) ? hn[d] : an[d - HDIM];
        const float* w = rwl + (size_t)d * NEXP;
#pragma unroll
        for (int e = 0; e < NEXP; ++e) acc[e] += v * w[e];
    }
    float cacc = 0.0f;
    const float* cwl = cW + (size_t)l * HDIM;
    for (int t = 0; t < 8; ++t) {
        int d = lane + t * 32;
        cacc += hn[d] * cwl[d];
    }
#pragma unroll
    for (int off = 16; off >= 1; off >>= 1) {
#pragma unroll
        for (int e = 0; e < NEXP; ++e) acc[e] += __shfl_xor(acc[e], off, 32);
        cacc += __shfl_xor(cacc, off, 32);
    }
    if (lane == 0) {
        float logit[NEXP];
#pragma unroll
        for (int e = 0; e < NEXP; ++e) logit[e] = acc[e] + rb[l * NEXP + e];
        int i0 = 0;
        for (int e = 1; e < NEXP; ++e) if (logit[e] > logit[i0]) i0 = e;
        int i1 = (i0 == 0) ? 1 : 0;
        for (int e = 0; e < NEXP; ++e) if (e != i0 && logit[e] > logit[i1]) i1 = e;
        float e1 = __expf(logit[i1] - logit[i0]);
        float g0 = 1.0f / (1.0f + e1);
        float g1 = e1 / (1.0f + e1);
        float conf = 1.0f / (1.0f + __expf(-(cacc + cb[l])));
        int p0 = atomicAdd(counts + i0, 1);
        idxl[(size_t)i0 * N + p0] = n; wl[(size_t)i0 * N + p0] = g0 * conf;
        int p1 = atomicAdd(counts + i1, 1);
        idxl[(size_t)i1 * N + p1] = n; wl[(size_t)i1 * N + p1] = g1 * conf;
    }
}

// ---------------------------------------------------------------------------
// dense input GEMM: h = x @ W_in + b_in   ([N,128] x [128,256])
// ---------------------------------------------------------------------------
__global__ __launch_bounds__(32) void gemm_in_kernel(const unsigned short* xb,
                                                     const unsigned short* fW,
                                                     const float* bias, float* h, int N) {
    int tile = blockIdx.x, lane = threadIdx.x;
    int kh = lane >> 4, nC = lane & 15;
    int rowA = tile * 16 + (lane & 15);
    if (rowA >= N) rowA = N - 1;
    const unsigned short* ar = xb + (size_t)rowA * INDIM;
    v16bf A[4];
#pragma unroll
    for (int kt = 0; kt < 4; ++kt)
        A[kt] = ld2(ar + kt * 32 + kh * 8, ar + kt * 32 + 16 + kh * 8);
    bool full = tile * 16 + 16 <= N;
    if (full) {
        float* hrow = h + (size_t)(tile * 16 + 8 * kh) * HDIM;
        for (int nt = 0; nt < 16; ++nt) {
            v8f c = {};
#pragma unroll
            for (int kt = 0; kt < 4; ++kt)
                c = wmma_bf16(A[kt], ldfrag(fW + (((size_t)kt * 16 + nt) * 32 + lane) * 16), c);
            int col = nt * 16 + nC;
            float b = bias[col];
#pragma unroll
            for (int r = 0; r < 8; ++r)
                hrow[(size_t)r * HDIM + col] = c[r] + b;
        }
    } else {
        for (int nt = 0; nt < 16; ++nt) {
            v8f c = {};
#pragma unroll
            for (int kt = 0; kt < 4; ++kt)
                c = wmma_bf16(A[kt], ldfrag(fW + (((size_t)kt * 16 + nt) * 32 + lane) * 16), c);
            int col = nt * 16 + nC;
            float b = bias[col];
#pragma unroll
            for (int r = 0; r < 8; ++r) {
                int row = tile * 16 + r + 8 * kh;
                if (row < N) h[(size_t)row * HDIM + col] = c[r] + b;
            }
        }
    }
}

// ---------------------------------------------------------------------------
// weak experts (e = 0..3): h[n] += w * relu(agg[n] @ Ww + bw)
// M=32 per wave: each B fragment feeds two WMMAs (halves L2 weight traffic)
// ---------------------------------------------------------------------------
__global__ __launch_bounds__(32) void weak_kernel(const unsigned short* aggb, float* h,
                                                  const unsigned short* fWeak, const float* wb,
                                                  const int* counts, const int* idxl,
                                                  const float* wl, int l, int N, int TILES32) {
    int e = blockIdx.x / TILES32, tile = blockIdx.x % TILES32;
    int cnt = counts[e];
    int base = tile * 32;
    if (base >= cnt) return;
    int lane = threadIdx.x, kh = lane >> 4, nC = lane & 15;
    const int* il = idxl + (size_t)e * N;
    const float* gl = wl + (size_t)e * N;
    int rA0 = base + (lane & 15);      if (rA0 >= cnt) rA0 = cnt - 1;
    int rA1 = base + 16 + (lane & 15); if (rA1 >= cnt) rA1 = cnt - 1;
    const unsigned short* ar0 = aggb + (size_t)il[rA0] * HDIM;
    const unsigned short* ar1 = aggb + (size_t)il[rA1] * HDIM;
    v16bf A0[8], A1[8];
#pragma unroll
    for (int kt = 0; kt < 8; ++kt) {
        A0[kt] = ld2(ar0 + kt * 32 + kh * 8, ar0 + kt * 32 + 16 + kh * 8);
        A1[kt] = ld2(ar1 + kt * 32 + kh * 8, ar1 + kt * 32 + 16 + kh * 8);
    }
    int nodeC0[8], nodeC1[8]; float wC0[8], wC1[8];
#pragma unroll
    for (int r = 0; r < 8; ++r) {
        int row0 = base + r + 8 * kh;
        int row1 = row0 + 16;
        nodeC0[r] = (row0 < cnt) ? il[row0] : 0; wC0[r] = (row0 < cnt) ? gl[row0] : 0.0f;
        nodeC1[r] = (row1 < cnt) ? il[row1] : 0; wC1[r] = (row1 < cnt) ? gl[row1] : 0.0f;
    }
    const unsigned short* fb = fWeak + (size_t)(l * 4 + e) * (HDIM * HDIM);
    const float* bb = wb + (size_t)(l * 4 + e) * HDIM;
    for (int nt = 0; nt < 16; ++nt) {
        v8f c0 = {}, c1 = {};
#pragma unroll
        for (int kt = 0; kt < 8; ++kt) {
            v16bf b = ldfrag(fb + (((size_t)kt * 16 + nt) * 32 + lane) * 16);
            c0 = wmma_bf16(A0[kt], b, c0);
            c1 = wmma_bf16(A1[kt], b, c1);
        }
        int col = nt * 16 + nC;
        float b = bb[col];
#pragma unroll
        for (int r = 0; r < 8; ++r) {
            float v0 = c0[r] + b; v0 = v0 > 0.0f ? v0 : 0.0f;
            float v1 = c1[r] + b; v1 = v1 > 0.0f ? v1 : 0.0f;
            atomicAdd(h + (size_t)nodeC0[r] * HDIM + col, wC0[r] * v0);
            atomicAdd(h + (size_t)nodeC1[r] * HDIM + col, wC1[r] * v1);
        }
    }
}

// ---------------------------------------------------------------------------
// strong experts (global e = 4..7): 2-layer MLP, 2x expansion.
// Stage-1 (16x512) output kept as bf16 in LDS, re-fragmented for stage 2.
// ---------------------------------------------------------------------------
__global__ __launch_bounds__(32) void strong_kernel(const unsigned short* aggb, float* h,
                                                    const unsigned short* fS1,
                                                    const unsigned short* fS2,
                                                    const float* b1, const float* b2,
                                                    const int* counts, const int* idxl,
                                                    const float* wl, int l, int N, int TILES) {
    __shared__ alignas(32) unsigned short sh[16 * 512];
    int e = blockIdx.x / TILES, tile = blockIdx.x % TILES;
    int ge = 4 + e;
    int cnt = counts[ge];
    int base = tile * 16;
    if (base >= cnt) return;
    int lane = threadIdx.x, kh = lane >> 4, nC = lane & 15;
    const int* il = idxl + (size_t)ge * N;
    const float* gl = wl + (size_t)ge * N;
    int rowA = base + (lane & 15);
    if (rowA >= cnt) rowA = cnt - 1;
    const unsigned short* ar = aggb + (size_t)il[rowA] * HDIM;
    v16bf A[8];
#pragma unroll
    for (int kt = 0; kt < 8; ++kt)
        A[kt] = ld2(ar + kt * 32 + kh * 8, ar + kt * 32 + 16 + kh * 8);
    int nodeC[8]; float wC[8];
#pragma unroll
    for (int r = 0; r < 8; ++r) {
        int row = base + r + 8 * kh;
        nodeC[r] = (row < cnt) ? il[row] : 0;
        wC[r] = (row < cnt) ? gl[row] : 0.0f;
    }
    const unsigned short* f1 = fS1 + (size_t)(l * 4 + e) * (HDIM * 2 * HDIM);
    const unsigned short* f2 = fS2 + (size_t)(l * 4 + e) * (2 * HDIM * HDIM);
    const float* bb1 = b1 + (size_t)(l * 4 + e) * (2 * HDIM);
    const float* bb2 = b2 + (size_t)(l * 4 + e) * HDIM;
    // stage 1: sh = relu(agg @ W1 + b1)
    for (int nt = 0; nt < 32; ++nt) {
        v8f c = {};
#pragma unroll
        for (int kt = 0; kt < 8; ++kt)
            c = wmma_bf16(A[kt], ldfrag(f1 + (((size_t)kt * 32 + nt) * 32 + lane) * 16), c);
        int col = nt * 16 + nC;
        float b = bb1[col];
#pragma unroll
        for (int r = 0; r < 8; ++r) {
            float v = c[r] + b;
            v = v > 0.0f ? v : 0.0f;
            sh[(size_t)(r + 8 * kh) * 512 + col] = f2bf(v);
        }
    }
    __syncthreads();
    // stage 2: h[n] += w * (sh @ W2 + b2)
    v16bf A2[16];
    const unsigned short* srow = sh + (size_t)(lane & 15) * 512;
#pragma unroll
    for (int kt = 0; kt < 16; ++kt)
        A2[kt] = ld2(srow + kt * 32 + kh * 8, srow + kt * 32 + 16 + kh * 8);
    for (int nt = 0; nt < 16; ++nt) {
        v8f c = {};
#pragma unroll
        for (int kt = 0; kt < 16; ++kt)
            c = wmma_bf16(A2[kt], ldfrag(f2 + (((size_t)kt * 16 + nt) * 32 + lane) * 16), c);
        int col = nt * 16 + nC;
        float b = bb2[col];
#pragma unroll
        for (int r = 0; r < 8; ++r)
            atomicAdd(h + (size_t)nodeC[r] * HDIM + col, wC[r] * (c[r] + b));
    }
}

// ---------------------------------------------------------------------------
// output GEMM: out = h @ W_out + b_out   ([N,256] x [256,128])
// ---------------------------------------------------------------------------
__global__ __launch_bounds__(32) void gemm_out_kernel(const unsigned short* hb,
                                                      const unsigned short* fW,
                                                      const float* bias, float* out, int N) {
    int tile = blockIdx.x, lane = threadIdx.x;
    int kh = lane >> 4, nC = lane & 15;
    int rowA = tile * 16 + (lane & 15);
    if (rowA >= N) rowA = N - 1;
    const unsigned short* ar = hb + (size_t)rowA * HDIM;
    v16bf A[8];
#pragma unroll
    for (int kt = 0; kt < 8; ++kt)
        A[kt] = ld2(ar + kt * 32 + kh * 8, ar + kt * 32 + 16 + kh * 8);
    bool full = tile * 16 + 16 <= N;
    if (full) {
        float* orow = out + (size_t)(tile * 16 + 8 * kh) * OUTDIM;
        for (int nt = 0; nt < 8; ++nt) {
            v8f c = {};
#pragma unroll
            for (int kt = 0; kt < 8; ++kt)
                c = wmma_bf16(A[kt], ldfrag(fW + (((size_t)kt * 8 + nt) * 32 + lane) * 16), c);
            int col = nt * 16 + nC;
            float b = bias[col];
#pragma unroll
            for (int r = 0; r < 8; ++r)
                orow[(size_t)r * OUTDIM + col] = c[r] + b;
        }
    } else {
        for (int nt = 0; nt < 8; ++nt) {
            v8f c = {};
#pragma unroll
            for (int kt = 0; kt < 8; ++kt)
                c = wmma_bf16(A[kt], ldfrag(fW + (((size_t)kt * 8 + nt) * 32 + lane) * 16), c);
            int col = nt * 16 + nC;
            float b = bias[col];
#pragma unroll
            for (int r = 0; r < 8; ++r) {
                int row = tile * 16 + r + 8 * kh;
                if (row < N) out[(size_t)row * OUTDIM + col] = c[r] + b;
            }
        }
    }
}

// ---------------------------------------------------------------------------
// host launcher
// ---------------------------------------------------------------------------
extern "C" void kernel_launch(void* const* d_in, const int* in_sizes, int n_in,
                              void* d_out, int out_size, void* d_ws, size_t ws_size,
                              hipStream_t stream) {
    const float*     x    = (const float*)d_in[0];
    const long long* ei   = (const long long*)d_in[1];
    const float*     W_in = (const float*)d_in[2];
    const float*     b_in = (const float*)d_in[3];
    const float*     rW   = (const float*)d_in[4];
    const float*     rb   = (const float*)d_in[5];
    const float*     wW   = (const float*)d_in[6];
    const float*     wb   = (const float*)d_in[7];
    const float*     sW1  = (const float*)d_in[8];
    const float*     sb1  = (const float*)d_in[9];
    const float*     sW2  = (const float*)d_in[10];
    const float*     sb2  = (const float*)d_in[11];
    const float*     cW   = (const float*)d_in[12];
    const float*     cb   = (const float*)d_in[13];
    const float*     W_out= (const float*)d_in[14];
    const float*     b_out= (const float*)d_in[15];
    float* out = (float*)d_out;

    const int N = in_sizes[0] / INDIM;       // 20000
    const int E = in_sizes[1] / 2;           // 320000
    const int TILES16 = (N + 15) / 16;
    const int TILES32 = (N + 31) / 32;

    // ---- workspace carve-out (256B aligned) ----
    char* basep = (char*)d_ws;
    size_t off = 0;
    auto carve = [&](size_t bytes) -> void* {
        off = (off + 255) & ~(size_t)255;
        void* p = basep + off;
        off += bytes;
        return p;
    };
    int*            degi   = (int*)carve((size_t)N * 4);
    int*            rowptr = (int*)carve((size_t)(N + 1) * 4);
    int*            fill   = (int*)carve((size_t)N * 4);
    int*            csr    = (int*)carve((size_t)E * 4);
    float*          agg    = (float*)carve((size_t)N * HDIM * 4);
    unsigned short* aggb   = (unsigned short*)carve((size_t)N * HDIM * 2);
    float*          hst    = (float*)carve((size_t)N * HDIM * 4);
    unsigned short* hb     = (unsigned short*)carve((size_t)N * HDIM * 2);
    unsigned short* xb     = (unsigned short*)carve((size_t)N * INDIM * 2);
    int*            counts = (int*)carve(NEXP * 4);
    int*            idxl   = (int*)carve((size_t)NEXP * N * 4);
    float*          wl     = (float*)carve((size_t)NEXP * N * 4);
    unsigned short* fWin   = (unsigned short*)carve((size_t)INDIM * HDIM * 2);
    unsigned short* fWeak  = (unsigned short*)carve((size_t)NLAYER * 4 * HDIM * HDIM * 2);
    unsigned short* fS1    = (unsigned short*)carve((size_t)NLAYER * 4 * HDIM * 2 * HDIM * 2);
    unsigned short* fS2    = (unsigned short*)carve((size_t)NLAYER * 4 * 2 * HDIM * HDIM * 2);
    unsigned short* fWout  = (unsigned short*)carve((size_t)HDIM * OUTDIM * 2);
    (void)ws_size; (void)n_in; (void)out_size;

    // ---- weight prep: f32 -> bf16 fragment swizzle ----
    cvt_bf16_kernel<<<(int)(((long long)N * INDIM + 255) / 256), 256, 0, stream>>>(x, xb, (long long)N * INDIM);
    swizzleB_kernel<<<1 * (INDIM / 32) * (HDIM / 16), 32, 0, stream>>>(W_in, fWin, INDIM, HDIM);
    swizzleB_kernel<<<12 * (HDIM / 32) * (HDIM / 16), 32, 0, stream>>>(wW, fWeak, HDIM, HDIM);
    swizzleB_kernel<<<12 * (HDIM / 32) * (2 * HDIM / 16), 32, 0, stream>>>(sW1, fS1, HDIM, 2 * HDIM);
    swizzleB_kernel<<<12 * (2 * HDIM / 32) * (HDIM / 16), 32, 0, stream>>>(sW2, fS2, 2 * HDIM, HDIM);
    swizzleB_kernel<<<1 * (HDIM / 32) * (OUTDIM / 16), 32, 0, stream>>>(W_out, fWout, HDIM, OUTDIM);

    // ---- CSR build (once; removes all per-layer aggregation atomics) ----
    zero_kernel<<<(N + 255) / 256, 256, 0, stream>>>((float*)degi, N);
    hist_kernel<<<(E + 255) / 256, 256, 0, stream>>>(ei, degi, E);
    scan_kernel<<<1, 256, 0, stream>>>(degi, rowptr, N);
    copy_int_kernel<<<(N + 255) / 256, 256, 0, stream>>>(rowptr, fill, N);
    build_csr_kernel<<<(E + 255) / 256, 256, 0, stream>>>(ei, fill, csr, E);

    // ---- input projection ----
    gemm_in_kernel<<<TILES16, 32, 0, stream>>>(xb, fWin, b_in, hst, N);

    // ---- layers ----
    for (int l = 0; l < NLAYER; ++l) {
        zero_kernel<<<1, 256, 0, stream>>>((float*)counts, NEXP);
        aggregate_kernel<<<(N + 7) / 8, 256, 0, stream>>>(hst, csr, rowptr, agg, aggb, N);
        router_kernel<<<(N + 7) / 8, 256, 0, stream>>>(hst, agg, rW, rb, cW, cb, counts, idxl, wl, l, N);
        weak_kernel<<<4 * TILES32, 32, 0, stream>>>(aggb, hst, fWeak, wb, counts, idxl, wl, l, N, TILES32);
        strong_kernel<<<4 * TILES16, 32, 0, stream>>>(aggb, hst, fS1, fS2, sb1, sb2, counts, idxl, wl, l, N, TILES16);
    }

    // ---- output projection ----
    cvt_bf16_kernel<<<(int)(((long long)N * HDIM + 255) / 256), 256, 0, stream>>>(hst, hb, (long long)N * HDIM);
    gemm_out_kernel<<<TILES16, 32, 0, stream>>>(hb, fWout, b_out, out, N);
}
